// Critic_8813272892044
// MI455X (gfx1250) — matrix operand order repaired and checked by reading, not compiled, not run on hardware
//
#include <hip/hip_runtime.h>

typedef __attribute__((ext_vector_type(16))) _Float16 v16h;
typedef __attribute__((ext_vector_type(8)))  _Float16 half8;
typedef __attribute__((ext_vector_type(8)))  float    v8f;
typedef __attribute__((ext_vector_type(4)))  int      i4;

#define BB 4096
#define VV 64
#define FF 15
#define HH 256

// CDNA5 async global->LDS path (guarded; falls back to sync staging)
#if defined(__AMDGCN__) && \
    __has_builtin(__builtin_amdgcn_global_load_async_to_lds_b128) && \
    __has_builtin(__builtin_amdgcn_s_wait_asynccnt)
#define USE_ASYNC_LDS 1
#else
#define USE_ASYNC_LDS 0
#endif

__device__ __forceinline__ void stage16(_Float16* dst, const _Float16* src) {
#if USE_ASYNC_LDS
    __builtin_amdgcn_global_load_async_to_lds_b128(
        (__attribute__((address_space(1))) i4*)(void*)const_cast<_Float16*>(src),
        (__attribute__((address_space(3))) i4*)(void*)dst,
        0, 0);
#else
    *(half8*)dst = *(const half8*)src;
#endif
}

// ---------------- utility kernels ----------------
__global__ void k_zero_f16(_Float16* p, int n) {
    int i = blockIdx.x * 256 + threadIdx.x;
    if (i < n) p[i] = (_Float16)0.f;
}
__global__ void k_zero_f32(float* p, int n) {
    int i = blockIdx.x * 256 + threadIdx.x;
    if (i < n) p[i] = 0.f;
}
// copy f32 (nrows x scols, row-major) -> f16 at dst[r*dstride + doff + c]
__global__ void k_cvt(_Float16* dst, int dstride, int doff,
                      const float* src, int scols, int nrows) {
    int i = blockIdx.x * 256 + threadIdx.x;
    if (i >= nrows * scols) return;
    int r = i / scols, c = i - r * scols;
    dst[(size_t)r * dstride + doff + c] = (_Float16)src[i];
}
__global__ void k_bias_add(float* dst, const float* a, const float* b, int n) {
    int i = blockIdx.x * 256 + threadIdx.x;
    if (i < n) dst[i] = a[i] + b[i];
}
// write x_t = [sv(15) | av(1)] into first 16 columns of combined input buffer
__global__ void k_pack(const float* __restrict__ state,
                       const float* __restrict__ action,
                       _Float16* __restrict__ XH, int Kp, int t) {
    int idx = blockIdx.x * 256 + threadIdx.x;
    if (idx >= BB * 16) return;
    int b = idx >> 4, c = idx & 15;
    float v = (c < FF) ? state[(size_t)b * (VV * FF) + t * FF + c]
                       : action[(size_t)b * VV + t];
    XH[(size_t)b * Kp + c] = (_Float16)v;
}

// ---------------- WMMA GEMM:  out[b,n] = act( sum_k A[b,k]*W[n,k] + bias[n] )
// MODE 0: f32 output (gates), MODE 1: ReLU -> f16 output
// 128x128 block tile, BK=32, double-buffered LDS, async global->LDS pipeline.
template <int MODE>
__global__ __launch_bounds__(256)
void k_gemm(const _Float16* __restrict__ A, int lda,
            const _Float16* __restrict__ W, int ldw,
            const float* __restrict__ bias,
            int Kp, int Ntot,
            float* __restrict__ outF, _Float16* __restrict__ outH, int ldo) {
    __shared__ __attribute__((aligned(16))) _Float16 As[2][128][40];
    __shared__ __attribute__((aligned(16))) _Float16 Ws[2][128][40];

    const int tid  = threadIdx.x;
    const int lane = tid & 31;          // wave32
    const int wid  = tid >> 5;          // 8 waves
    const int wm   = wid & 1;           // 2 wave-rows  (64 rows each)
    const int wn   = wid >> 1;          // 4 wave-cols  (32 cols each)
    const int m0   = blockIdx.y * 128;
    const int n0   = blockIdx.x * 128;

    v8f zero;
    #pragma unroll
    for (int e = 0; e < 8; ++e) zero[e] = 0.f;
    v8f acc[4][2];
    #pragma unroll
    for (int i = 0; i < 4; ++i)
        #pragma unroll
        for (int j = 0; j < 2; ++j) acc[i][j] = zero;

    // stage one 128x32 K-tile of A and W into LDS buffer `buf`
    auto stage = [&](int buf, int k0) {
        #pragma unroll
        for (int it = 0; it < 2; ++it) {
            int idx = tid + it * 256;          // 0..511
            int r = idx >> 2, cc = idx & 3;    // row, 8-half (16B) chunk
            const _Float16* ga = A + (size_t)(m0 + r) * lda + k0 + cc * 8;
            const _Float16* gw = W + (size_t)(n0 + r) * ldw + k0 + cc * 8;
            stage16(&As[buf][r][cc * 8], ga);
            stage16(&Ws[buf][r][cc * 8], gw);
#if !USE_ASYNC_LDS
            __builtin_prefetch(ga + 32, 0, 0);  // gfx1250 global_prefetch_b8
            __builtin_prefetch(gw + 32, 0, 0);
#endif
        }
    };

    const int nk = Kp >> 5;
    stage(0, 0);
    for (int ik = 0; ik < nk; ++ik) {
        const int cur = ik & 1;
        if (ik + 1 < nk) {
            stage(cur ^ 1, (ik + 1) * 32);      // prefetch next tile into other buffer
#if USE_ASYNC_LDS
            __builtin_amdgcn_s_wait_asynccnt(4); // keep only the 4 just-issued in flight
#endif
        } else {
#if USE_ASYNC_LDS
            __builtin_amdgcn_s_wait_asynccnt(0);
#endif
        }
        __syncthreads();

        // build fragments per documented wave32 layouts
        const int akb = (lane >> 4) * 8;   // A: lanes 16-31 hold K=8..15 / 24..31
        const int bkb = (lane >> 4) * 8;   // B: lanes 16-31 hold K=8..15 / 24..31
        v16h af[4], bf[2];
        #pragma unroll
        for (int i = 0; i < 4; ++i) {
            int r = wm * 64 + i * 16 + (lane & 15);
            half8 lo = *(const half8*)&As[cur][r][akb];
            half8 hi = *(const half8*)&As[cur][r][akb + 16];
            af[i] = __builtin_shufflevector(lo, hi, 0,1,2,3,4,5,6,7,8,9,10,11,12,13,14,15);
        }
        #pragma unroll
        for (int j = 0; j < 2; ++j) {
            int n = wn * 32 + j * 16 + (lane & 15);
            half8 lo = *(const half8*)&Ws[cur][n][bkb];
            half8 hi = *(const half8*)&Ws[cur][n][bkb + 16];
            bf[j] = __builtin_shufflevector(lo, hi, 0,1,2,3,4,5,6,7,8,9,10,11,12,13,14,15);
        }

        #pragma unroll
        for (int i = 0; i < 4; ++i)
            #pragma unroll
            for (int j = 0; j < 2; ++j)
                acc[i][j] = __builtin_amdgcn_wmma_f32_16x16x32_f16(
                    false, af[i], false, bf[j], (short)0, acc[i][j], false, false);
        __syncthreads();   // all waves done reading `cur` before it is re-staged
    }

    // epilogue: C/D layout — lanes 0-15: M=v, N=lane; lanes 16-31: M=8+v, N=lane-16
    const int nl   = lane & 15;
    const int rsel = (lane >> 4) * 8;
    #pragma unroll
    for (int i = 0; i < 4; ++i) {
        #pragma unroll
        for (int j = 0; j < 2; ++j) {
            int col = n0 + wn * 32 + j * 16 + nl;
            float bv = bias[col];
            #pragma unroll
            for (int v = 0; v < 8; ++v) {
                int row = m0 + wm * 64 + i * 16 + rsel + v;
                float val = acc[i][j][v] + bv;
                if (MODE == 0) {
                    outF[(size_t)row * ldo + col] = val;
                } else {
                    val = val > 0.f ? val : 0.f;
                    outH[(size_t)row * ldo + col] = (_Float16)val;
                }
            }
        }
    }
}

// ---------------- LSTM cell update (+ optional q head, block = one batch row)
__global__ __launch_bounds__(256)
void k_lstm_cell(const float* __restrict__ G, float* __restrict__ C,
                 _Float16* __restrict__ XH, int Kp, int hoff,
                 const float* __restrict__ qw, const float* __restrict__ qb,
                 float* __restrict__ qout, int t) {
    __shared__ float red[256];
    const int b = blockIdx.x;
    const int j = threadIdx.x;
    const float* g = G + (size_t)b * 1024;
    float gi = g[j], gf = g[256 + j], gg = g[512 + j], go = g[768 + j];
    float si = 1.f / (1.f + __expf(-gi));
    float sf = 1.f / (1.f + __expf(-gf));
    float so = 1.f / (1.f + __expf(-go));
    float c  = sf * C[(size_t)b * HH + j] + si * tanhf(gg);
    C[(size_t)b * HH + j] = c;
    float h = so * tanhf(c);
    XH[(size_t)b * Kp + hoff + j] = (_Float16)h;
    if (qw != nullptr) {
        red[j] = h * qw[j];
        __syncthreads();
        #pragma unroll
        for (int s = 128; s > 0; s >>= 1) {
            if (j < s) red[j] += red[j + s];
            __syncthreads();
        }
        if (j == 0) qout[(size_t)b * VV + t] = red[0] + qb[0];
    }
}

// ---------------- host orchestration ----------------
static inline int gup(int n) { return (n + 255) / 256; }

extern "C" void kernel_launch(void* const* d_in, const int* in_sizes, int n_in,
                              void* d_out, int out_size, void* d_ws, size_t ws_size,
                              hipStream_t stream) {
    const float* state  = (const float*)d_in[0];
    const float* action = (const float*)d_in[1];

    char* p = (char*)d_ws;
    auto alloc = [&](size_t bytes) -> void* {
        void* r = (void*)p;
        p += (bytes + 255) & ~(size_t)255;
        return r;
    };
    _Float16* W1c  = (_Float16*)alloc((size_t)1024 * 288 * 2);   // [Wih1|Whh1|pad]
    _Float16* W2c  = (_Float16*)alloc((size_t)1024 * 544 * 2);   // [Wih2|Whh2|pad]
    _Float16* Wf1  = (_Float16*)alloc((size_t)1024 * 256 * 2);
    _Float16* Wf2  = (_Float16*)alloc((size_t)1024 * 1024 * 2);
    _Float16* Wf3  = (_Float16*)alloc((size_t)512 * 1024 * 2);
    _Float16* Wf4  = (_Float16*)alloc((size_t)256 * 512 * 2);
    float*    bc1  = (float*)alloc(1024 * 4);
    float*    bc2  = (float*)alloc(1024 * 4);
    _Float16* xh1  = (_Float16*)alloc((size_t)BB * 288 * 2);     // [x(16)|h(256)|pad16]
    _Float16* xh2  = (_Float16*)alloc((size_t)BB * 544 * 2);     // [x(16)|mlp(256)|h(256)|pad16]
    _Float16* act1 = (_Float16*)alloc((size_t)BB * 1024 * 2);
    _Float16* act2 = (_Float16*)alloc((size_t)BB * 1024 * 2);
    _Float16* act3 = (_Float16*)alloc((size_t)BB * 512 * 2);
    float*    gts  = (float*)alloc((size_t)BB * 1024 * 4);
    float*    cbuf = (float*)alloc((size_t)BB * 256 * 4);

    const dim3 blk(256);
    const dim3 gN1024(1024 / 128, BB / 128);
    const dim3 gN512(512 / 128, BB / 128);
    const dim3 gN256(256 / 128, BB / 128);

    for (int twin = 0; twin < 2; ++twin) {
        const int base = 2 + twin * 18;
        const float* Wih1 = (const float*)d_in[base + 0];
        const float* bih1 = (const float*)d_in[base + 1];
        const float* Whh1 = (const float*)d_in[base + 2];
        const float* bhh1 = (const float*)d_in[base + 3];
        const float* f1w  = (const float*)d_in[base + 4];
        const float* f1b  = (const float*)d_in[base + 5];
        const float* f2w  = (const float*)d_in[base + 6];
        const float* f2b  = (const float*)d_in[base + 7];
        const float* f3w  = (const float*)d_in[base + 8];
        const float* f3b  = (const float*)d_in[base + 9];
        const float* f4w  = (const float*)d_in[base + 10];
        const float* f4b  = (const float*)d_in[base + 11];
        const float* Wih2 = (const float*)d_in[base + 12];
        const float* bih2 = (const float*)d_in[base + 13];
        const float* Whh2 = (const float*)d_in[base + 14];
        const float* bhh2 = (const float*)d_in[base + 15];
        const float* qw   = (const float*)d_in[base + 16];
        const float* qb   = (const float*)d_in[base + 17];
        float* qout = (float*)d_out + (size_t)twin * BB * VV;

        // ---- weight prep (f16, combined, zero-padded K) ----
        k_zero_f16<<<gup(1024 * 288), blk, 0, stream>>>(W1c, 1024 * 288);
        k_zero_f16<<<gup(1024 * 544), blk, 0, stream>>>(W2c, 1024 * 544);
        k_cvt<<<gup(1024 * 16), blk, 0, stream>>>(W1c, 288, 0, Wih1, 16, 1024);
        k_cvt<<<gup(1024 * 256), blk, 0, stream>>>(W1c, 288, 16, Whh1, 256, 1024);
        k_cvt<<<gup(1024 * 272), blk, 0, stream>>>(W2c, 544, 0, Wih2, 272, 1024);
        k_cvt<<<gup(1024 * 256), blk, 0, stream>>>(W2c, 544, 272, Whh2, 256, 1024);
        k_cvt<<<gup(1024 * 256), blk, 0, stream>>>(Wf1, 256, 0, f1w, 256, 1024);
        k_cvt<<<gup(1024 * 1024), blk, 0, stream>>>(Wf2, 1024, 0, f2w, 1024, 1024);
        k_cvt<<<gup(512 * 1024), blk, 0, stream>>>(Wf3, 1024, 0, f3w, 1024, 512);
        k_cvt<<<gup(256 * 512), blk, 0, stream>>>(Wf4, 512, 0, f4w, 512, 256);
        k_bias_add<<<gup(1024), blk, 0, stream>>>(bc1, bih1, bhh1, 1024);
        k_bias_add<<<gup(1024), blk, 0, stream>>>(bc2, bih2, bhh2, 1024);

        // ---- LSTM1: h,c = 0 ----
        k_zero_f16<<<gup(BB * 288), blk, 0, stream>>>(xh1, BB * 288);
        k_zero_f32<<<gup(BB * 256), blk, 0, stream>>>(cbuf, BB * 256);
        for (int t = 0; t < VV; ++t) {
            k_pack<<<gup(BB * 16), blk, 0, stream>>>(state, action, xh1, 288, t);
            k_gemm<0><<<gN1024, blk, 0, stream>>>(xh1, 288, W1c, 288, bc1,
                                                  288, 1024, gts, nullptr, 1024);
            k_lstm_cell<<<BB, blk, 0, stream>>>(gts, cbuf, xh1, 288, 16,
                                                nullptr, nullptr, nullptr, 0);
        }

        // ---- MLP (fc4 writes straight into xh2 broadcast columns) ----
        k_zero_f16<<<gup(BB * 544), blk, 0, stream>>>(xh2, BB * 544);
        k_gemm<1><<<gN1024, blk, 0, stream>>>(xh1 + 16, 288, Wf1, 256, f1b,
                                              256, 1024, nullptr, act1, 1024);
        k_gemm<1><<<gN1024, blk, 0, stream>>>(act1, 1024, Wf2, 1024, f2b,
                                              1024, 1024, nullptr, act2, 1024);
        k_gemm<1><<<gN512, blk, 0, stream>>>(act2, 1024, Wf3, 1024, f3b,
                                             1024, 512, nullptr, act3, 512);
        k_gemm<1><<<gN256, blk, 0, stream>>>(act3, 512, Wf4, 512, f4b,
                                             512, 256, nullptr, xh2 + 16, 544);

        // ---- LSTM2 + q head ----
        k_zero_f32<<<gup(BB * 256), blk, 0, stream>>>(cbuf, BB * 256);
        for (int t = 0; t < VV; ++t) {
            k_pack<<<gup(BB * 16), blk, 0, stream>>>(state, action, xh2, 544, t);
            k_gemm<0><<<gN1024, blk, 0, stream>>>(xh2, 544, W2c, 544, bc2,
                                                  544, 1024, gts, nullptr, 1024);
            k_lstm_cell<<<BB, blk, 0, stream>>>(gts, cbuf, xh2, 544, 272,
                                                qw, qb, qout, t);
        }
    }
}
